// net_gcn_84524956385822
// MI455X (gfx1250) — compile-verified, matrix operand order checked by reading, not run on hardware
//
#include <hip/hip_runtime.h>
#include <math.h>

typedef __bf16 bf16;
typedef __attribute__((ext_vector_type(16))) __bf16 v16bf;
typedef __attribute__((ext_vector_type(8)))  float  v8f;

// ---------------------------------------------------------------------------
// Elementwise conversion f32 -> bf16
// ---------------------------------------------------------------------------
__global__ void k_f32_to_bf16(const float* __restrict__ in, bf16* __restrict__ out, long n) {
  long i = (long)blockIdx.x * blockDim.x + threadIdx.x;
  if (i < n) out[i] = (bf16)in[i];
}

// Transpose W[K,Nin] (f32) -> Bt[Npad,K] (bf16), zero-padding columns >= Nin.
__global__ void k_prep_bt(const float* __restrict__ W, bf16* __restrict__ Bt,
                          int K, int Nin, int Npad) {
  long i = (long)blockIdx.x * blockDim.x + threadIdx.x;
  long total = (long)Npad * K;
  if (i >= total) return;
  int n = (int)(i / K);
  int k = (int)(i % K);
  Bt[i] = (bf16)((n < Nin) ? W[(long)k * Nin + n] : 0.0f);
}

__global__ void k_fill(float* __restrict__ p, float v, long n) {
  long i = (long)blockIdx.x * blockDim.x + threadIdx.x;
  if (i < n) p[i] = v;
}

// ---------------------------------------------------------------------------
// WMMA bf16 GEMM: C[M,N] = act(A[M,K] @ Bt^T + bias)
//   A  : [M,K] bf16 row-major ; Bt : [N,K] bf16 row-major (W transposed)
// One wave computes a (16*MT) x (16*NT) tile: MT A-fragments are shared across
// NT B-fragments -> MT*NT v_wmma_f32_16x16x32_bf16 per K-step with only
// (2*MT + 2*NT) b128 loads. All per-lane addresses are 32-bit byte offsets off
// the uniform kernarg base (saddr+voffset form, no per-lane 64-bit adds).
// Fragment layout per ISA 7.12.2 (16-bit A 16x32): lane = m | (khalf<<4),
// elements 0..7 -> K = k0 + 8*khalf .. +7 ; elements 8..15 -> +16.
// ---------------------------------------------------------------------------
template <int MT, int NT, int ACT>
__global__ void wmma_gemm(const bf16* __restrict__ A, const bf16* __restrict__ Bt,
                          const float* __restrict__ bias, int bias_n,
                          float* __restrict__ C, bf16* __restrict__ Cb,
                          int M, int K, int N) {
  const int lane  = threadIdx.x & 31;
  const int wave  = blockIdx.x * (blockDim.x >> 5) + (threadIdx.x >> 5);
  const int mg    = M / (16 * MT);       // row groups (M multiple of 16*MT)
  const int ng    = N / (16 * NT);       // col groups
  if (wave >= mg * ng) return;           // wave-uniform; EXEC stays all-ones
  const int tm    = wave / ng;
  const int tg    = wave % ng;
  const int khalf = lane >> 4;
  const int ml    = lane & 15;

  const char* Ab = (const char*)A;
  const char* Bb = (const char*)Bt;

  unsigned aoff[MT], boff[NT];
#pragma unroll
  for (int r = 0; r < MT; ++r)
    aoff[r] = (unsigned)((((tm * MT + r) * 16 + ml) * K + 8 * khalf) * 2);
#pragma unroll
  for (int t = 0; t < NT; ++t)
    boff[t] = (unsigned)((((tg * NT + t) * 16 + ml) * K + 8 * khalf) * 2);

  v8f acc[MT][NT] = {};

#pragma unroll 2
  for (int k0 = 0; k0 < K; k0 += 32) {
    union { v16bf v; uint4 u[2]; } af[MT];
#pragma unroll
    for (int r = 0; r < MT; ++r) {
      af[r].u[0] = *(const uint4*)(Ab + (aoff[r] + 2 * k0));
      af[r].u[1] = *(const uint4*)(Ab + (aoff[r] + 2 * k0 + 32));
    }
#pragma unroll
    for (int t = 0; t < NT; ++t) {
      union { v16bf v; uint4 u[2]; } bb;
      bb.u[0] = *(const uint4*)(Bb + (boff[t] + 2 * k0));
      bb.u[1] = *(const uint4*)(Bb + (boff[t] + 2 * k0 + 32));
#pragma unroll
      for (int r = 0; r < MT; ++r)
        acc[r][t] = __builtin_amdgcn_wmma_f32_16x16x32_bf16(
            /*neg_a=*/false, af[r].v, /*neg_b=*/false, bb.v,
            /*c_mod=*/(short)0, acc[r][t], /*reuse_a=*/false, /*reuse_b=*/false);
    }
  }

#pragma unroll
  for (int r = 0; r < MT; ++r) {
#pragma unroll
    for (int t = 0; t < NT; ++t) {
      const int nn = (tg * NT + t) * 16 + ml;
      float bv = 0.0f;
      if (bias) bv = (nn < bias_n) ? bias[nn] : 0.0f;
#pragma unroll
      for (int v = 0; v < 8; ++v) {
        const int mm = (tm * MT + r) * 16 + v + 8 * khalf;  // C layout: VGPR v, lanes>=16 -> M+8
        float val = acc[r][t][v] + bv;
        if (ACT) val = fmaxf(val, 0.0f);
        if (C)  C[(long)mm * N + nn]  = val;
        if (Cb) Cb[(long)mm * N + nn] = (bf16)val;
      }
    }
  }
}

// ---------------------------------------------------------------------------
// Soft edge weight pipeline
// ---------------------------------------------------------------------------
// q[i,c] = sum_d logits[i,d] * relu(2*P0[d,c]);  logits stride 16, q stride 8
__global__ void k_compute_q(const float* __restrict__ logits, const float* __restrict__ p0,
                            float* __restrict__ q, long Nn) {
  long i = (long)blockIdx.x * blockDim.x + threadIdx.x;
  if (i >= Nn * 7) return;
  long nidx = i / 7;
  int  c    = (int)(i % 7);
  float s = 0.0f;
#pragma unroll
  for (int d = 0; d < 7; ++d)
    s += logits[nidx * 16 + d] * fmaxf(2.0f * p0[d * 7 + c], 0.0f);
  q[nidx * 8 + c] = s;
}

// raw[e] = dot(logits[row[e]], q[col[e]]) over 7 classes
__global__ void k_ew_raw(const float* __restrict__ logits, const float* __restrict__ q,
                         const int* __restrict__ row, const int* __restrict__ col,
                         float* __restrict__ raw, long E) {
  long e = (long)blockIdx.x * blockDim.x + threadIdx.x;
  if (e >= E) return;
  long r = row[e], c = col[e];
  float s = 0.0f;
#pragma unroll
  for (int d = 0; d < 7; ++d) s += logits[r * 16 + d] * q[c * 8 + d];
  raw[e] = s;
}

__global__ void k_reduce(const float* __restrict__ raw, long n, float* __restrict__ red) {
  __shared__ float s1[256], s2[256];
  float a = 0.0f, b = 0.0f;
  for (long i = (long)blockIdx.x * blockDim.x + threadIdx.x; i < n;
       i += (long)gridDim.x * blockDim.x) {
    float v = raw[i];
    a += v; b += v * v;
  }
  s1[threadIdx.x] = a; s2[threadIdx.x] = b;
  __syncthreads();
  for (int o = 128; o > 0; o >>= 1) {
    if ((int)threadIdx.x < o) {
      s1[threadIdx.x] += s1[threadIdx.x + o];
      s2[threadIdx.x] += s2[threadIdx.x + o];
    }
    __syncthreads();
  }
  if (threadIdx.x == 0) {
    unsafeAtomicAdd(&red[0], s1[0]);
    unsafeAtomicAdd(&red[1], s2[0]);
  }
}

__global__ void k_stats(float* __restrict__ red, float Ef) {
  float mean = red[0] / Ef;
  float var  = (red[1] - Ef * mean * mean) / (Ef - 1.0f);   // unbiased (ddof=1)
  red[2] = mean;
  red[3] = sqrtf(1e-4f / var);
}

// ew[e] = (raw-mean)*scale + 1 ; deg[col] += ew (deg pre-filled with 1.0 self-loop)
__global__ void k_ew_norm(const float* __restrict__ raw, const float* __restrict__ red,
                          const int* __restrict__ col, float* __restrict__ ew,
                          float* __restrict__ deg, long E) {
  long e = (long)blockIdx.x * blockDim.x + threadIdx.x;
  if (e >= E) return;
  float w = (raw[e] - red[2]) * red[3] + 1.0f;
  ew[e] = w;
  unsafeAtomicAdd(&deg[col[e]], w);
}

__global__ void k_dinv(const float* __restrict__ deg, float* __restrict__ dinv, long n) {
  long i = (long)blockIdx.x * blockDim.x + threadIdx.x;
  if (i >= n) return;
  float d = deg[i];
  dinv[i] = (d > 0.0f) ? rsqrtf(fmaxf(d, 1e-12f)) : 0.0f;
}

__global__ void k_edge_norm(const float* __restrict__ ew, const float* __restrict__ dinv,
                            const int* __restrict__ row, const int* __restrict__ col,
                            float* __restrict__ nrm, long E) {
  long e = (long)blockIdx.x * blockDim.x + threadIdx.x;
  if (e >= E) return;
  nrm[e] = dinv[row[e]] * ew[e] * dinv[col[e]];
}

// ---------------------------------------------------------------------------
// Edge scatter: agg[col,:] += h[row,:] * norm[e].  DIM/4 threads per edge,
// float4 reads + 4 f32 global atomics per thread (global_atomic_add_f32).
// ---------------------------------------------------------------------------
template <int DIM>
__global__ void edge_scatter(const float* __restrict__ h, const float* __restrict__ nrm,
                             const int* __restrict__ row, const int* __restrict__ col,
                             float* __restrict__ agg, long E) {
  constexpr int TPE = DIM / 4;
  long gid = (long)blockIdx.x * blockDim.x + threadIdx.x;
  long e   = gid / TPE;
  int  t   = (int)(gid % TPE);
  if (e >= E) return;
  long r = row[e], c = col[e];
  float w = nrm[e];
  float4 hv = *(const float4*)(h + r * DIM + t * 4);
  float* dst = agg + c * DIM + t * 4;
  unsafeAtomicAdd(dst + 0, hv.x * w);
  unsafeAtomicAdd(dst + 1, hv.y * w);
  unsafeAtomicAdd(dst + 2, hv.z * w);
  unsafeAtomicAdd(dst + 3, hv.w * w);
}

// ---------------------------------------------------------------------------
// GCN finalize variants (self-loop term + bias [+ activation / residual])
// ---------------------------------------------------------------------------
// layer 0: h0 = agg + h*dinv^2 + b ; x1 = relu(h0) (bf16)
__global__ void k_fin0(const float* __restrict__ agg, const float* __restrict__ h,
                       const float* __restrict__ dinv, const float* __restrict__ bias,
                       float* __restrict__ h0, bf16* __restrict__ x1, long total, int dim) {
  long i = (long)blockIdx.x * blockDim.x + threadIdx.x;
  if (i >= total) return;
  long nidx = i / dim;
  int  d    = (int)(i % dim);
  float di = dinv[nidx];
  float v  = agg[i] + h[i] * di * di + bias[d];
  h0[i] = v;
  x1[i] = (bf16)fmaxf(v, 0.0f);
}

// layer 1: x2 = relu(agg + h*dinv^2 + b) + h0 (bf16)
__global__ void k_fin1(const float* __restrict__ agg, const float* __restrict__ h,
                       const float* __restrict__ dinv, const float* __restrict__ bias,
                       const float* __restrict__ h0, bf16* __restrict__ x2,
                       long total, int dim) {
  long i = (long)blockIdx.x * blockDim.x + threadIdx.x;
  if (i >= total) return;
  long nidx = i / dim;
  int  d    = (int)(i % dim);
  float di = dinv[nidx];
  float v  = agg[i] + h[i] * di * di + bias[d];
  x2[i] = (bf16)(fmaxf(v, 0.0f) + h0[i]);
}

// layer 2 (padded dim 16 -> out dim 7): out = agg + h*dinv^2 + b
__global__ void k_fin2(const float* __restrict__ agg, const float* __restrict__ h,
                       const float* __restrict__ dinv, const float* __restrict__ bias,
                       float* __restrict__ out, long Nn) {
  long i = (long)blockIdx.x * blockDim.x + threadIdx.x;
  if (i >= Nn * 7) return;
  long nidx = i / 7;
  int  d    = (int)(i % 7);
  float di = dinv[nidx];
  out[i] = agg[nidx * 16 + d] + h[nidx * 16 + d] * di * di + bias[d];
}

// ---------------------------------------------------------------------------
extern "C" void kernel_launch(void* const* d_in, const int* in_sizes, int n_in,
                              void* d_out, int out_size, void* d_ws, size_t ws_size,
                              hipStream_t stream) {
  const float* x   = (const float*)d_in[0];
  const int*   ei  = (const int*)d_in[1];
  const float* W1  = (const float*)d_in[2];  const float* b1  = (const float*)d_in[3];
  const float* W2  = (const float*)d_in[4];  const float* b2  = (const float*)d_in[5];
  const float* W3  = (const float*)d_in[6];  const float* b3  = (const float*)d_in[7];
  const float* mw1 = (const float*)d_in[8];  const float* mb1 = (const float*)d_in[9];
  const float* mw2 = (const float*)d_in[10]; const float* mb2 = (const float*)d_in[11];
  const float* mw3 = (const float*)d_in[12]; const float* mb3 = (const float*)d_in[13];
  const float* p0  = (const float*)d_in[14];
  float* out = (float*)d_out;

  const long Nn   = (long)in_sizes[0] / 512;     // 50000 (multiple of 16)
  const long Ee   = (long)in_sizes[1] / 2;       // 800000
  const long Mpad = (Nn + 31) & ~31L;            // 50016: rows padded for MT=2
  const int* row = ei;
  const int* col = ei + Ee;

  // ---- workspace carve-out (256B aligned, deterministic) ----
  char* wsb = (char*)d_ws;
  size_t cur = 0;
  auto alloc = [&](size_t bytes) -> char* {
    char* p = wsb + cur;
    cur = (cur + bytes + 255) & ~(size_t)255;
    return p;
  };
  bf16*  x_bf   = (bf16*)alloc(Mpad * 512 * 2);
  bf16*  mw1t   = (bf16*)alloc(512L * 512 * 2);
  bf16*  mw2t   = (bf16*)alloc(64L * 512 * 2);
  bf16*  mw3t   = (bf16*)alloc(16L * 64 * 2);
  bf16*  g1t    = (bf16*)alloc(256L * 512 * 2);
  bf16*  g2t    = (bf16*)alloc(256L * 256 * 2);
  bf16*  g3t    = (bf16*)alloc(16L * 256 * 2);
  bf16*  h1_bf  = (bf16*)alloc(Mpad * 512 * 2);
  bf16*  h2_bf  = (bf16*)alloc(Mpad * 64 * 2);
  float* logits = (float*)alloc(Mpad * 16 * 4);
  float* q      = (float*)alloc(Nn * 8 * 4);
  float* ewraw  = (float*)alloc(Ee * 4);
  float* ew     = (float*)alloc(Ee * 4);
  float* nrm    = (float*)alloc(Ee * 4);
  float* deg    = (float*)alloc(Nn * 4);
  float* dinv   = (float*)alloc(Nn * 4);
  float* red    = (float*)alloc(256);
  float* hbuf   = (float*)alloc(Mpad * 256 * 4);
  float* agg    = (float*)alloc(Nn * 256 * 4);
  float* h0     = (float*)alloc(Nn * 256 * 4);
  bf16*  x1_bf  = (bf16*)alloc(Mpad * 256 * 2);
  bf16*  x2_bf  = (bf16*)alloc(Mpad * 256 * 2);

  auto cdiv = [](long a, long b) -> unsigned { return (unsigned)((a + b - 1) / b); };

  // ---- zero the row-padding of wave-fed activation buffers (tiny) ----
  hipMemsetAsync(x_bf  + Nn * 512, 0, (Mpad - Nn) * 512 * 2, stream);
  hipMemsetAsync(x1_bf + Nn * 256, 0, (Mpad - Nn) * 256 * 2, stream);
  hipMemsetAsync(x2_bf + Nn * 256, 0, (Mpad - Nn) * 256 * 2, stream);

  // ---- precision prep: activations + transposed/padded weights in bf16 ----
  k_f32_to_bf16<<<cdiv(Nn * 512, 256), 256, 0, stream>>>(x, x_bf, Nn * 512);
  k_prep_bt<<<cdiv(512L * 512, 256), 256, 0, stream>>>(mw1, mw1t, 512, 512, 512);
  k_prep_bt<<<cdiv(64L * 512, 256), 256, 0, stream>>>(mw2, mw2t, 512, 64, 64);
  k_prep_bt<<<cdiv(16L * 64, 256), 256, 0, stream>>>(mw3, mw3t, 64, 7, 16);
  k_prep_bt<<<cdiv(256L * 512, 256), 256, 0, stream>>>(W1, g1t, 512, 256, 256);
  k_prep_bt<<<cdiv(256L * 256, 256), 256, 0, stream>>>(W2, g2t, 256, 256, 256);
  k_prep_bt<<<cdiv(16L * 256, 256), 256, 0, stream>>>(W3, g3t, 256, 7, 16);

  // ---- pseudo MLP (all WMMA, 32x64 wave tiles) ----
  { long waves = (Mpad / 32) * (512 / 64);
    wmma_gemm<2, 4, 1><<<cdiv(waves, 8), 256, 0, stream>>>(
        x_bf, mw1t, mb1, 512, nullptr, h1_bf, (int)Mpad, 512, 512); }
  { long waves = (Mpad / 32) * (64 / 64);
    wmma_gemm<2, 4, 1><<<cdiv(waves, 8), 256, 0, stream>>>(
        h1_bf, mw2t, mb2, 64, nullptr, h2_bf, (int)Mpad, 512, 64); }
  { long waves = (Mpad / 32);
    wmma_gemm<2, 1, 0><<<cdiv(waves, 8), 256, 0, stream>>>(
        h2_bf, mw3t, mb3, 7, logits, nullptr, (int)Mpad, 64, 16); }

  // ---- learned soft edge weights ----
  k_compute_q<<<cdiv(Nn * 7, 256), 256, 0, stream>>>(logits, p0, q, Nn);
  k_ew_raw<<<cdiv(Ee, 256), 256, 0, stream>>>(logits, q, row, col, ewraw, Ee);
  hipMemsetAsync(red, 0, 16, stream);
  k_reduce<<<512, 256, 0, stream>>>(ewraw, Ee, red);
  k_stats<<<1, 1, 0, stream>>>(red, (float)Ee);
  k_fill<<<cdiv(Nn, 256), 256, 0, stream>>>(deg, 1.0f, Nn);   // self-loop weight
  k_ew_norm<<<cdiv(Ee, 256), 256, 0, stream>>>(ewraw, red, col, ew, deg, Ee);
  k_dinv<<<cdiv(Nn, 256), 256, 0, stream>>>(deg, dinv, Nn);
  k_edge_norm<<<cdiv(Ee, 256), 256, 0, stream>>>(ew, dinv, row, col, nrm, Ee);

  // ---- GCN layer 0: h = x@W1 ; h0 = agg + self + b1 ; x1 = relu(h0) ----
  { long waves = (Mpad / 32) * (256 / 64);
    wmma_gemm<2, 4, 0><<<cdiv(waves, 8), 256, 0, stream>>>(
        x_bf, g1t, nullptr, 0, hbuf, nullptr, (int)Mpad, 512, 256); }
  hipMemsetAsync(agg, 0, Nn * 256 * 4, stream);
  edge_scatter<256><<<cdiv(Ee * 64, 256), 256, 0, stream>>>(hbuf, nrm, row, col, agg, Ee);
  k_fin0<<<cdiv(Nn * 256, 256), 256, 0, stream>>>(agg, hbuf, dinv, b1, h0, x1_bf, Nn * 256, 256);

  // ---- GCN layer 1: x2 = relu(gcn(x1)) + h0 ----
  { long waves = (Mpad / 32) * (256 / 64);
    wmma_gemm<2, 4, 0><<<cdiv(waves, 8), 256, 0, stream>>>(
        x1_bf, g2t, nullptr, 0, hbuf, nullptr, (int)Mpad, 256, 256); }
  hipMemsetAsync(agg, 0, Nn * 256 * 4, stream);
  edge_scatter<256><<<cdiv(Ee * 64, 256), 256, 0, stream>>>(hbuf, nrm, row, col, agg, Ee);
  k_fin1<<<cdiv(Nn * 256, 256), 256, 0, stream>>>(agg, hbuf, dinv, b2, h0, x2_bf, Nn * 256, 256);

  // ---- GCN layer 2 (padded N=16, emit 7 cols) ----
  { long waves = (Mpad / 32);
    wmma_gemm<2, 1, 0><<<cdiv(waves, 8), 256, 0, stream>>>(
        x2_bf, g3t, nullptr, 0, hbuf, nullptr, (int)Mpad, 256, 16); }
  hipMemsetAsync(agg, 0, Nn * 16 * 4, stream);
  edge_scatter<16><<<cdiv(Ee * 4, 256), 256, 0, stream>>>(hbuf, nrm, row, col, agg, Ee);
  k_fin2<<<cdiv(Nn * 7, 256), 256, 0, stream>>>(agg, hbuf, dinv, b3, out, Nn);
}
